// ModuleSelector_25864293056980
// MI455X (gfx1250) — matrix operand order repaired and checked by reading, not compiled, not run on hardware
//
#include <hip/hip_runtime.h>

#define NUM_FEATS   65536
#define IN_SIZE     1024
#define OUT_SIZE    1024
#define NUM_MODULES 8

#define BM      128
#define BN      128
#define BK      32
#define LDT     40                 // padded bf16 row stride in LDS (80B: 16B-aligned, bank-rotating)
#define THREADS 256
#define KSTEPS  (IN_SIZE / BK)     // 32

typedef __attribute__((ext_vector_type(16))) __bf16 v16bf;
typedef __attribute__((ext_vector_type(8)))  __bf16 v8bf;
typedef __attribute__((ext_vector_type(2)))  __bf16 v2bf;
typedef __attribute__((ext_vector_type(8)))  float  v8f;

union Frag { v16bf v; v8bf h[2]; };

// Native float -> bf16 conversion: lowers to v_cvt_pk_bf16_f32 on gfx1250.
__device__ inline __bf16 f2bf(float f) { return (__bf16)f; }

// ---------------- binning (counting sort of row ids by expert) ----------------

__global__ void moe_init(int* meta) {
    if (threadIdx.x < 24) meta[threadIdx.x] = 0;   // counts[8], cursor[8], offsets[8]
}

__global__ void moe_count(const int* __restrict__ ids, int* __restrict__ counts) {
    int i = blockIdx.x * blockDim.x + threadIdx.x;
    if (i < NUM_FEATS) atomicAdd(&counts[ids[i] & 7], 1);
}

__global__ void moe_scan(const int* __restrict__ counts, int* __restrict__ offsets) {
    if (threadIdx.x == 0) {
        int s = 0;
        for (int e = 0; e < NUM_MODULES; ++e) { offsets[e] = s; s += counts[e]; }
    }
}

__global__ void moe_scatter(const int* __restrict__ ids, const int* __restrict__ offsets,
                            int* __restrict__ cursor, int* __restrict__ perm) {
    int i = blockIdx.x * blockDim.x + threadIdx.x;
    if (i < NUM_FEATS) {
        int e = ids[i] & 7;
        int slot = offsets[e] + atomicAdd(&cursor[e], 1);
        perm[slot] = i;
    }
}

// ---------------- grouped GEMM: Y[perm rows of expert e] = bf16(X) @ bf16(W[e]) + b[e] ----

__global__ __launch_bounds__(THREADS)
void moe_gemm(const float* __restrict__ X, const float* __restrict__ W,
              const float* __restrict__ bias,
              const int* __restrict__ counts, const int* __restrict__ offsets,
              const int* __restrict__ perm, float* __restrict__ Y)
{
    const int e   = blockIdx.z;
    const int cnt = counts[e];
    const int m0  = blockIdx.y * BM;
    if (m0 >= cnt) return;                       // uniform early-exit for unused tiles
    const int n0   = blockIdx.x * BN;
    const int base = offsets[e];

    __shared__ __bf16 As[2][BM * LDT];
    __shared__ __bf16 Bs[2][BN * LDT];
    __shared__ int rows[BM];

    const int tid = threadIdx.x;
    if (tid < BM) {
        int i = m0 + tid;
        rows[tid] = (i < cnt) ? perm[base + i] : -1;
    }
    __syncthreads();

    // --- A loader: 4 rows/thread, 4 consecutive k each (fixed across k-steps) ---
    const int ar = tid >> 3;                     // row group 0..31 (x4 passes)
    const int ac = (tid & 7) * 4;                // k offset 0,4,..,28
    const float* aPtr[4];
#pragma unroll
    for (int p = 0; p < 4; ++p) {
        int row = rows[p * 32 + ar];
        if (row < 0) row = 0;                    // padding rows: load row 0, store is masked later
        aPtr[p] = X + (size_t)row * IN_SIZE + ac;
    }
    // --- B loader: k-pairs so transposed LDS stores pack as b32 ---
    const int bkq = (tid >> 5) * 2;              // k pair base 0,2,..,14 (+16 on pass 1)
    const int bn  = (tid & 31) * 4;              // n offset 0..124
    const float* bPtr = W + (size_t)e * IN_SIZE * OUT_SIZE + (size_t)bkq * OUT_SIZE + n0 + bn;

    // --- wave tiling: 8 waves as 4(M) x 2(N); each wave 32x64 = 2x4 WMMA frags ---
    const int wave  = tid >> 5;
    const int lane  = tid & 31;
    const int lr    = lane & 15;
    const int lh    = lane >> 4;
    const int waveM = (wave & 3) * 32;
    const int waveN = (wave >> 2) * 64;

    v8f acc[2][4];
#pragma unroll
    for (int mi = 0; mi < 2; ++mi)
#pragma unroll
        for (int ni = 0; ni < 4; ++ni)
#pragma unroll
            for (int r = 0; r < 8; ++r) acc[mi][ni][r] = 0.0f;

    float4 ra[4], rb[4];
    auto loadGlobal = [&](int kk) {              // issue global loads (no LDS yet)
#pragma unroll
        for (int p = 0; p < 4; ++p)
            ra[p] = *(const float4*)(aPtr[p] + kk);
        const float* bp = bPtr + (size_t)kk * OUT_SIZE;
#pragma unroll
        for (int p = 0; p < 2; ++p) {            // rows bkq+p*16 and bkq+p*16+1
            rb[2 * p]     = *(const float4*)(bp + (size_t)(p * 16) * OUT_SIZE);
            rb[2 * p + 1] = *(const float4*)(bp + (size_t)(p * 16 + 1) * OUT_SIZE);
        }
    };
    auto storeLds = [&](int buf) {               // convert fp32->bf16 and stage
        __bf16* a = &As[buf][0];
#pragma unroll
        for (int p = 0; p < 4; ++p) {
            __bf16* d = a + (p * 32 + ar) * LDT + ac;
            d[0] = f2bf(ra[p].x); d[1] = f2bf(ra[p].y);
            d[2] = f2bf(ra[p].z); d[3] = f2bf(ra[p].w);
        }
        __bf16* b = &Bs[buf][0];
#pragma unroll
        for (int p = 0; p < 2; ++p) {            // transpose: Bs is [n][k]; pack (k,k+1) per n
            int k = bkq + p * 16;
            const float4& lo = rb[2 * p];
            const float4& hi = rb[2 * p + 1];
            *(v2bf*)(b + (bn + 0) * LDT + k) = v2bf{f2bf(lo.x), f2bf(hi.x)};
            *(v2bf*)(b + (bn + 1) * LDT + k) = v2bf{f2bf(lo.y), f2bf(hi.y)};
            *(v2bf*)(b + (bn + 2) * LDT + k) = v2bf{f2bf(lo.z), f2bf(hi.z)};
            *(v2bf*)(b + (bn + 3) * LDT + k) = v2bf{f2bf(lo.w), f2bf(hi.w)};
        }
    };

    loadGlobal(0);
    storeLds(0);

    for (int step = 0; step < KSTEPS; ++step) {
        const int buf = step & 1;
        if (step + 1 < KSTEPS) loadGlobal((step + 1) * BK);   // overlap with compute
        __syncthreads();

        Frag afr[2], bfr[4];
        const __bf16* a = &As[buf][0];
#pragma unroll
        for (int mi = 0; mi < 2; ++mi) {
            // A 16-bit 16x32: elems 0..7 -> k = lh*8+j ; elems 8..15 -> k = 16+lh*8+j
            const __bf16* src = a + (waveM + mi * 16 + lr) * LDT;
            afr[mi].h[0] = *(const v8bf*)(src + lh * 8);
            afr[mi].h[1] = *(const v8bf*)(src + 16 + lh * 8);
        }
        const __bf16* b = &Bs[buf][0];
#pragma unroll
        for (int ni = 0; ni < 4; ++ni) {
            // B 16-bit 32x16: elems 0..7 -> k = lh*8+j ; elems 8..15 -> k = 16+lh*8+j (Bs[n][k])
            const __bf16* src = b + (waveN + ni * 16 + lr) * LDT;
            bfr[ni].h[0] = *(const v8bf*)(src + lh * 8);
            bfr[ni].h[1] = *(const v8bf*)(src + 16 + lh * 8);
        }
#pragma unroll
        for (int mi = 0; mi < 2; ++mi)
#pragma unroll
            for (int ni = 0; ni < 4; ++ni)
                acc[mi][ni] = __builtin_amdgcn_wmma_f32_16x16x32_bf16(
                    false, afr[mi].v, false, bfr[ni].v,
                    (short)0, acc[mi][ni], false, false);

        if (step + 1 < KSTEPS) storeLds(buf ^ 1);  // barrier at loop top protects both buffers
    }

    // --- epilogue: D layout lane(lr,lh): vgpr r -> m = r + lh*8, n = lr ---
#pragma unroll
    for (int ni = 0; ni < 4; ++ni) {
        int nCol = n0 + waveN + ni * 16 + lr;
        float bv = bias[e * OUT_SIZE + nCol];
#pragma unroll
        for (int mi = 0; mi < 2; ++mi) {
#pragma unroll
            for (int r = 0; r < 8; ++r) {
                int mRow = waveM + mi * 16 + lh * 8 + r;
                int row  = rows[mRow];
                if (row >= 0)
                    Y[(size_t)row * OUT_SIZE + nCol] = acc[mi][ni][r] + bv;
            }
        }
    }
}

// ---------------- launch ----------------

extern "C" void kernel_launch(void* const* d_in, const int* in_sizes, int n_in,
                              void* d_out, int out_size, void* d_ws, size_t ws_size,
                              hipStream_t stream) {
    const float* X    = (const float*)d_in[0];
    const int*   ids  = (const int*)d_in[1];      // integer input -> int32 per harness convention
    const float* W    = (const float*)d_in[2];
    const float* bias = (const float*)d_in[3];
    float*       Y    = (float*)d_out;

    int* meta    = (int*)d_ws;                    // counts[8] | cursor[8] | offsets[8]
    int* counts  = meta;
    int* cursor  = meta + 8;
    int* offsets = meta + 16;
    int* perm    = (int*)((char*)d_ws + 256);     // 65536 ints = 256 KB

    moe_init   <<<1, 32, 0, stream>>>(meta);
    moe_count  <<<NUM_FEATS / 256, 256, 0, stream>>>(ids, counts);
    moe_scan   <<<1, 32, 0, stream>>>(counts, offsets);
    moe_scatter<<<NUM_FEATS / 256, 256, 0, stream>>>(ids, offsets, cursor, perm);

    dim3 grid(OUT_SIZE / BN, NUM_FEATS / BM, NUM_MODULES);   // 8 x 512 x 8; extras exit early
    moe_gemm<<<grid, THREADS, 0, stream>>>(X, W, bias, counts, offsets, perm, Y);
}
